// GraphAttentionEncoder_75617194213833
// MI455X (gfx1250) — matrix operand order repaired and checked by reading, not compiled, not run on hardware
//
#include <hip/hip_runtime.h>

typedef __attribute__((ext_vector_type(16))) _Float16 v16h;
typedef __attribute__((ext_vector_type(8)))  float    v8f;
typedef __attribute__((ext_vector_type(8)))  _Float16 h8;
typedef __attribute__((ext_vector_type(4)))  _Float16 h4;
typedef __attribute__((ext_vector_type(4)))  int      v4i;

union AF { v16h v; _Float16 h[16]; };
union CF { v8f  v; float    f[8];  };

#define AS1 __attribute__((address_space(1)))
#define AS3 __attribute__((address_space(3)))

#if __has_builtin(__builtin_amdgcn_global_load_async_to_lds_b128)
#define HAVE_ASYNC 1
#else
#define HAVE_ASYNC 0
#endif

__device__ __forceinline__ void async_fence() {
#if HAVE_ASYNC
#if __has_builtin(__builtin_amdgcn_s_wait_asynccnt)
  __builtin_amdgcn_s_wait_asynccnt(0);
#else
  asm volatile("s_wait_asynccnt 0" ::: "memory");
#endif
#endif
  asm volatile("" ::: "memory");
}

// Stage a 32-row x width-halfs f16 tile (row stride rowStride halfs) into per-wave LDS.
// width must be a multiple of 8; no divergence (total is a multiple of 32).
__device__ __forceinline__ void stage_tile(const _Float16* __restrict__ gsrc, int rowStride,
                                           _Float16* __restrict__ ldst, int width, int lane) {
  int blks = width >> 3;           // 16B blocks per row
  int total = 32 * blks;
  for (int q = lane; q < total; q += 32) {
    int row = q / blks, cb = (q % blks) << 3;
    const _Float16* gp = gsrc + (size_t)row * rowStride + cb;
    _Float16* lp = ldst + row * width + cb;
#if HAVE_ASYNC
    __builtin_amdgcn_global_load_async_to_lds_b128(
        (AS1 v4i*)(uintptr_t)gp, (AS3 v4i*)(uintptr_t)lp, 0, 0);
#else
    *(h8*)lp = *(const h8*)gp;
#endif
  }
  async_fence();
}

__device__ __forceinline__ v8f wmma_f32_f16(v16h a, v16h b, v8f c) {
  // D = A(16x32 f16) * B(32x16 f16) + C(16x16 f32)
  return __builtin_amdgcn_wmma_f32_16x16x32_f16(false, a, false, b, (short)0, c, false, false);
}
// A-fragment half index j -> K offset (contiguous order: k = 16*hi + j, matching B fills)
__device__ __forceinline__ int a_kmap(int hi, int j) { return hi * 16 + j; }
__device__ __forceinline__ float celu1(float x) { return x > 0.f ? x : (__expf(x) - 1.f); }

// ------------------------- LayerNorm (wave per row) -------------------------
__global__ void ln_rows_k(const float* __restrict__ x, float* __restrict__ y,
                          const float* __restrict__ g, const float* __restrict__ b,
                          int rows, int D, int act) {
  int w = (int)((blockIdx.x * blockDim.x + threadIdx.x) >> 5);
  int lane = threadIdx.x & 31;
  if (w >= rows) return;
  const float* xr = x + (size_t)w * D;
  float s = 0.f, q = 0.f;
  for (int d = lane; d < D; d += 32) { float v = xr[d]; s += v; q += v * v; }
  for (int o = 16; o > 0; o >>= 1) { s += __shfl_xor(s, o); q += __shfl_xor(q, o); }
  float mu = s / (float)D;
  float var = q / (float)D - mu * mu;
  float rs = rsqrtf(var + 1e-5f);
  float* yr = y + (size_t)w * D;
  for (int d = lane; d < D; d += 32) {
    float v = (xr[d] - mu) * rs * g[d] + b[d];
    if (act == 3) v = 0.5f * v * (1.f + erff(v * 0.70710678118654752f));  // exact gelu
    yr[d] = v;
  }
}

// ------------------------- Generic WMMA GEMM: Y = act(X @ W^T + bias [+ res]) -------------------------
#define GB_M 64
#define GB_N 64
#define GB_K 32
__global__ void gemm_k(const float* __restrict__ X, const float* __restrict__ W,
                       const float* __restrict__ bias, const float* __restrict__ res,
                       float* __restrict__ Y, int M, int K, int Nout, int act, int addres) {
  __shared__ _Float16 As[GB_M][GB_K + 8];
  __shared__ _Float16 Bs[GB_N][GB_K + 8];
  int tid = threadIdx.x;
  int lane = tid & 31, wv = tid >> 5;
  int rowBase = blockIdx.x * GB_M;
  int colBase = blockIdx.y * GB_N;
  int mr = lane & 15, hi = lane >> 4;
  bool alnA = ((M & (GB_M - 1)) == 0) && ((K & 31) == 0);      // uniform
  bool alnB = ((Nout & (GB_N - 1)) == 0) && ((K & 31) == 0);   // uniform
  CF acc[4];
  for (int t = 0; t < 4; t++) for (int r = 0; r < 8; r++) acc[t].f[r] = 0.f;

  for (int k0 = 0; k0 < K; k0 += GB_K) {
    if (alnA) {
      for (int i = tid; i < GB_M * (GB_K / 4); i += blockDim.x) {
        int r = i >> 3, kq = (i & 7) * 4;
        const float4 v = *(const float4*)(X + (size_t)(rowBase + r) * K + k0 + kq);
        h4 hv = {(_Float16)v.x, (_Float16)v.y, (_Float16)v.z, (_Float16)v.w};
        *(h4*)(&As[r][kq]) = hv;
      }
    } else {
      for (int i = tid; i < GB_M * GB_K; i += blockDim.x) {
        int r = i >> 5, k = i & 31;
        int gm = rowBase + r, gk = k0 + k;
        As[r][k] = (gm < M && gk < K) ? (_Float16)X[(size_t)gm * K + gk] : (_Float16)0.f;
      }
    }
    if (alnB) {
      for (int i = tid; i < GB_N * (GB_K / 4); i += blockDim.x) {
        int r = i >> 3, kq = (i & 7) * 4;
        const float4 v = *(const float4*)(W + (size_t)(colBase + r) * K + k0 + kq);
        h4 hv = {(_Float16)v.x, (_Float16)v.y, (_Float16)v.z, (_Float16)v.w};
        *(h4*)(&Bs[r][kq]) = hv;
      }
    } else {
      for (int i = tid; i < GB_N * GB_K; i += blockDim.x) {
        int c = i >> 5, k = i & 31;
        int gc = colBase + c, gk = k0 + k;
        Bs[c][k] = (gc < Nout && gk < K) ? (_Float16)W[(size_t)gc * K + gk] : (_Float16)0.f;
      }
    }
    __syncthreads();
    AF a;
    int arow = wv * 16 + mr;
    for (int j = 0; j < 16; j++) a.h[j] = As[arow][a_kmap(hi, j)];
    for (int t = 0; t < 4; t++) {
      AF bf;
      int bcol = t * 16 + mr;
      for (int j = 0; j < 16; j++) bf.h[j] = Bs[bcol][hi * 16 + j];
      acc[t].v = wmma_f32_f16(a.v, bf.v, acc[t].v);
    }
    __syncthreads();
  }
  for (int t = 0; t < 4; t++) {
    int c = colBase + t * 16 + mr;
    if (c >= Nout) continue;
    float bb = bias ? bias[c] : 0.f;
    for (int r = 0; r < 8; r++) {
      int m = rowBase + wv * 16 + r + 8 * hi;
      if (m >= M) continue;
      float v = acc[t].f[r] + bb;
      if (addres) v += res[(size_t)m * Nout + c];
      if (act == 1) v = celu1(v);
      else if (act == 2) v = (v >= 0.f) ? v : 0.2f * v;
      Y[(size_t)m * Nout + c] = v;
    }
  }
}

// ------------------------- Fused edge MLP (per 16-row tile of flattened BNN x E) -------------------------
// e_out = LN(edge2(celu(edge1(LN(e)))) + e) ; ee = e_out . a_edge
__global__ void edge_fused_k(const float* __restrict__ e_in, float* __restrict__ e_out,
                             float* __restrict__ ee,
                             const float* __restrict__ ln2g, const float* __restrict__ ln2b,
                             const float* __restrict__ w1, const float* __restrict__ b1,
                             const float* __restrict__ w2, const float* __restrict__ b2,
                             const float* __restrict__ log_, const float* __restrict__ lob_,
                             const float* __restrict__ asum, int aeo) {
  __shared__ _Float16 As[4][16][32];
  __shared__ _Float16 Hs[4][16][32];
  int lane = threadIdx.x & 31, wv = threadIdx.x >> 5;
  size_t base = ((size_t)blockIdx.x * 4 + wv) * 16;
  int mr = lane & 15, hi = lane >> 4;
  if (lane < 16) {
    const float* row = e_in + (base + lane) * 16;
    float v[16]; float s = 0.f, q = 0.f;
    for (int k = 0; k < 16; k++) { v[k] = row[k]; s += v[k]; q += v[k] * v[k]; }
    float mu = s * 0.0625f, var = q * 0.0625f - mu * mu, rs = rsqrtf(var + 1e-5f);
    for (int k = 0; k < 16; k++) As[wv][lane][k] = (_Float16)((v[k] - mu) * rs * ln2g[k] + ln2b[k]);
    for (int k = 16; k < 32; k++) As[wv][lane][k] = (_Float16)0.f;
  }
  __syncthreads();
  AF a;
  for (int j = 0; j < 16; j++) a.h[j] = As[wv][mr][a_kmap(hi, j)];
  for (int t = 0; t < 2; t++) {                 // edge1: E(16, zero-padded K) -> 2E(32)
    AF bf;
    int col = t * 16 + mr;
    for (int j = 0; j < 16; j++) {
      int k = hi * 16 + j;
      bf.h[j] = (k < 16) ? (_Float16)w1[col * 16 + k] : (_Float16)0.f;
    }
    CF h; for (int r = 0; r < 8; r++) h.f[r] = 0.f;
    h.v = wmma_f32_f16(a.v, bf.v, h.v);
    float bb = b1[col];
    for (int r = 0; r < 8; r++) Hs[wv][r + 8 * hi][col] = (_Float16)celu1(h.f[r] + bb);
  }
  __syncthreads();
  AF a2, b2f;                                    // edge2: 2E(32) -> E(16)
  for (int j = 0; j < 16; j++) a2.h[j] = Hs[wv][mr][a_kmap(hi, j)];
  for (int j = 0; j < 16; j++) b2f.h[j] = (_Float16)w2[mr * 32 + hi * 16 + j];
  CF d; for (int r = 0; r < 8; r++) d.f[r] = 0.f;
  d.v = wmma_f32_f16(a2.v, b2f.v, d.v);
  float gg = log_[mr], bo = lob_[mr], ae = asum[aeo + mr], bb2 = b2[mr];
  for (int r = 0; r < 8; r++) {
    size_t rowi = base + r + 8 * hi;
    float val = d.f[r] + bb2 + e_in[rowi * 16 + mr];   // +residual
    float s = val, q = val * val;                      // LN across 16 lanes of half-wave
    for (int o = 1; o < 16; o <<= 1) { s += __shfl_xor(s, o); q += __shfl_xor(q, o); }
    float mu = s * 0.0625f, var = q * 0.0625f - mu * mu, rs = rsqrtf(var + 1e-5f);
    float out = (val - mu) * rs * gg + bo;
    e_out[rowi * 16 + mr] = out;
    float ev = out * ae;
    for (int o = 1; o < 16; o <<= 1) ev += __shfl_xor(ev, o);
    if (mr == 0) ee[rowi] = ev;
  }
}

// ------------------------- small helpers -------------------------
__global__ void asum_k(const float* __restrict__ A, float* __restrict__ asum, int heads, int D) {
  int d = blockIdx.x * blockDim.x + threadIdx.x;
  if (d >= D) return;
  float s = 0.f;
  for (int h = 0; h < heads; h++) s += A[h * D + d];
  asum[d] = s;
}
__global__ void rc_k(const float* __restrict__ node, const float* __restrict__ asum,
                     float* __restrict__ r, float* __restrict__ c,
                     int BN, int heads, int hs, int outf) {
  int i = blockIdx.x * blockDim.x + threadIdx.x;
  if (i >= BN * heads) return;
  int n = i / heads, h = i % heads;
  const float* nd = node + (size_t)n * outf + h * hs;
  float sr = 0.f, sc = 0.f;
  for (int d = 0; d < hs; d++) { sr += nd[d] * asum[d]; sc += nd[d] * asum[hs + d]; }
  r[i] = sr; c[i] = sc;
}
__global__ void cvt_h_k(const float* __restrict__ x, _Float16* __restrict__ y, size_t n) {
  size_t i = (size_t)blockIdx.x * blockDim.x + threadIdx.x;
  size_t st = (size_t)gridDim.x * blockDim.x;
  for (; i < n; i += st) y[i] = (_Float16)x[i];
}
__global__ void add3_k(const float* __restrict__ res, const float* __restrict__ x2,
                       const float* __restrict__ g, const float* __restrict__ gamma,
                       float* __restrict__ out, size_t n) {
  float gm = gamma[0];
  size_t i = (size_t)blockIdx.x * blockDim.x + threadIdx.x;
  size_t st = (size_t)gridDim.x * blockDim.x;
  for (; i < n; i += st) out[i] = res[i] + x2[i] + gm * g[i];
}

__device__ __forceinline__ void softmax_rows16(float* P, int lane, int ncols) {
  for (int i = 0; i < 16; i++) {
    float* row = P + i * ncols;
    float mx = -3.0e38f;
    for (int m = lane; m < ncols; m += 32) mx = fmaxf(mx, row[m]);
    for (int o = 16; o > 0; o >>= 1) mx = fmaxf(mx, __shfl_xor(mx, o));
    float s = 0.f;
    for (int m = lane; m < ncols; m += 32) { float e = __expf(row[m] - mx); row[m] = e; s += e; }
    for (int o = 16; o > 0; o >>= 1) s += __shfl_xor(s, o);
    float inv = 1.f / s;
    for (int m = lane; m < ncols; m += 32) row[m] *= inv;
  }
}

// ------------------------- GAT attention + message (flash-style, LDS probs, async V staging) -------------------------
__global__ void gat_attn_msg_k(const _Float16* __restrict__ node_h,
                               const float* __restrict__ r, const float* __restrict__ c,
                               const float* __restrict__ ee, const int* __restrict__ adj,
                               float* __restrict__ msg,
                               int N_, int heads, int hs, int outf) {
  __shared__ float P[2][16 * 512];
  __shared__ __align__(16) _Float16 Vt[2][32 * 128];
  int lane = threadIdx.x & 31, wv = threadIdx.x >> 5;
  int gw = blockIdx.x * 2 + wv;
  int tiles = N_ / 16;
  int bh = gw / tiles, t = gw % tiles;
  int bi = bh / heads, h = bh % heads;
  int n0 = t * 16;
  float* Pr = P[wv];
  _Float16* vt = Vt[wv];
  int mr = lane & 15, hi = lane >> 4;

  float rv[16];
  for (int i = 0; i < 16; i++) rv[i] = r[(bi * N_ + n0 + i) * heads + h];
  for (int m = lane; m < N_; m += 32) {
    float cv = c[(bi * N_ + m) * heads + h];
    for (int i = 0; i < 16; i++) {
      size_t idx = ((size_t)(bi * N_ + n0 + i)) * N_ + m;
      float lg = rv[i] + cv + ee[idx];
      if (adj[idx] == 0) lg = -1.0e9f;
      lg = (lg >= 0.f) ? lg : 0.2f * lg;        // leaky after mask
      lg = fminf(fmaxf(lg, -1.0e9f), 1.0e9f);
      Pr[i * N_ + m] = lg;
    }
  }
  __syncthreads();
  softmax_rows16(Pr, lane, N_);
  __syncthreads();
  int dt = hs / 16;
  CF acc[8];
  for (int ct = 0; ct < dt; ct++) for (int q = 0; q < 8; q++) acc[ct].f[q] = 0.f;
  const _Float16* vsrc = node_h + (size_t)bi * N_ * outf + h * hs;
  for (int kc = 0; kc < N_; kc += 32) {
    stage_tile(vsrc + (size_t)kc * outf, outf, vt, hs, lane);   // 32 x hs tile of node_h
    AF a;
    for (int j = 0; j < 16; j++) a.h[j] = (_Float16)Pr[mr * N_ + kc + a_kmap(hi, j)];
    for (int ct = 0; ct < dt; ct++) {
      AF bf;
      for (int j = 0; j < 16; j++) bf.h[j] = vt[(hi * 16 + j) * hs + ct * 16 + mr];
      acc[ct].v = wmma_f32_f16(a.v, bf.v, acc[ct].v);
    }
  }
  for (int ct = 0; ct < dt; ct++) {
    int col = h * hs + ct * 16 + mr;
    for (int q = 0; q < 8; q++)
      msg[((size_t)(bi * N_ + n0 + q + 8 * hi)) * outf + col] = acc[ct].f[q];
  }
}

// ------------------------- MHA (QK^T via WMMA, softmax in LDS, PV via WMMA, async K/V staging) -------------------------
__global__ void mha_attn_k(const _Float16* __restrict__ qkv_h, float* __restrict__ o,
                           int N_, int heads, int hd, int outf) {
  __shared__ float P[2][16 * 512];
  __shared__ __align__(16) _Float16 Vt[2][32 * 128];
  int lane = threadIdx.x & 31, wv = threadIdx.x >> 5;
  int gw = blockIdx.x * 2 + wv;
  int tiles = N_ / 16;
  int bh = gw / tiles, t = gw % tiles;
  int bi = bh / heads, h = bh % heads;
  int n0 = t * 16;
  float* Pr = P[wv];
  _Float16* vt = Vt[wv];
  int mr = lane & 15, hi = lane >> 4;
  int s3 = 3 * outf;
  int kch = hd / 32;
  AF aq[4];
  for (int kc = 0; kc < kch; kc++)
    for (int j = 0; j < 16; j++)
      aq[kc].h[j] = qkv_h[((size_t)(bi * N_ + n0 + mr)) * s3 + h * hd + kc * 32 + a_kmap(hi, j)];
  float scale = rsqrtf((float)hd);
  const _Float16* ksrc = qkv_h + (size_t)bi * N_ * s3 + outf + h * hd;
  for (int mt = 0; mt < N_; mt += 32) {
    stage_tile(ksrc + (size_t)mt * s3, s3, vt, hd, lane);       // 32 x hd tile of K
    for (int half = 0; half < 2; half++) {
      CF acc; for (int q = 0; q < 8; q++) acc.f[q] = 0.f;
      int rbase = half * 16 + mr;
      for (int kc = 0; kc < kch; kc++) {
        AF bf;
        for (int j = 0; j < 16; j++) bf.h[j] = vt[rbase * hd + kc * 32 + hi * 16 + j];
        acc.v = wmma_f32_f16(aq[kc].v, bf.v, acc.v);
      }
      for (int q = 0; q < 8; q++)
        Pr[(q + 8 * hi) * N_ + mt + half * 16 + mr] = acc.f[q] * scale;
    }
  }
  __syncthreads();
  softmax_rows16(Pr, lane, N_);
  __syncthreads();
  int dt = hd / 16;
  CF acc2[8];
  for (int ct = 0; ct < dt; ct++) for (int q = 0; q < 8; q++) acc2[ct].f[q] = 0.f;
  const _Float16* vvsrc = qkv_h + (size_t)bi * N_ * s3 + 2 * outf + h * hd;
  for (int kc = 0; kc < N_; kc += 32) {
    stage_tile(vvsrc + (size_t)kc * s3, s3, vt, hd, lane);      // 32 x hd tile of V
    AF a;
    for (int j = 0; j < 16; j++) a.h[j] = (_Float16)Pr[mr * N_ + kc + a_kmap(hi, j)];
    for (int ct = 0; ct < dt; ct++) {
      AF bf;
      for (int j = 0; j < 16; j++) bf.h[j] = vt[(hi * 16 + j) * hd + ct * 16 + mr];
      acc2[ct].v = wmma_f32_f16(a.v, bf.v, acc2[ct].v);
    }
  }
  for (int ct = 0; ct < dt; ct++) {
    int dcol = h * hd + ct * 16 + mr;
    for (int q = 0; q < 8; q++)
      o[((size_t)(bi * N_ + n0 + q + 8 * hi)) * outf + dcol] = acc2[ct].f[q];
  }
}

// ------------------------- pooling -------------------------
__global__ void pool_score_k(const float* __restrict__ x, const float* __restrict__ w,
                             const float* __restrict__ b, float* __restrict__ s, int BN, int F) {
  int i = blockIdx.x * blockDim.x + threadIdx.x;
  if (i >= BN) return;
  const float* xr = x + (size_t)i * F;
  float acc = b[0];
  for (int k = 0; k < F; k++) acc += xr[k] * w[k];
  s[i] = acc;
}
__global__ void pool_softmax_k(float* __restrict__ s, int N_) {
  int bi = blockIdx.x, lane = threadIdx.x;
  float* sr = s + bi * N_;
  float mx = -3.0e38f;
  for (int n = lane; n < N_; n += 32) mx = fmaxf(mx, sr[n]);
  for (int o = 16; o > 0; o >>= 1) mx = fmaxf(mx, __shfl_xor(mx, o));
  float sm = 0.f;
  for (int n = lane; n < N_; n += 32) { float e = __expf(sr[n] - mx); sr[n] = e; sm += e; }
  for (int o = 16; o > 0; o >>= 1) sm += __shfl_xor(sm, o);
  float inv = 1.f / sm;
  for (int n = lane; n < N_; n += 32) sr[n] *= inv;
}
__global__ void pool_weighted_k(const float* __restrict__ s, const float* __restrict__ x,
                                float* __restrict__ pooled, int B_, int N_, int F) {
  int i = blockIdx.x * blockDim.x + threadIdx.x;
  if (i >= B_ * F) return;
  int bi = i / F, f = i % F;
  float acc = 0.f;
  for (int n = 0; n < N_; n++) acc += s[bi * N_ + n] * x[((size_t)bi * N_ + n) * F + f];
  pooled[i] = acc;
}

// ------------------------- host orchestration -------------------------
extern "C" void kernel_launch(void* const* d_in, const int* in_sizes, int n_in,
                              void* d_out, int out_size, void* d_ws, size_t ws_size,
                              hipStream_t stream) {
  (void)n_in; (void)out_size; (void)ws_size;
  const int B_ = 8, N_ = 512, IN_ = 64, HID_ = 128, OUTF = 128, E_ = 16, NL = 3;
  const int HEADS_[3] = {4, 4, 1};
  const int BN = B_ * N_;
  const size_t BNN = (size_t)B_ * N_ * N_;

  // array inputs: detect sorted (adj, edge, node) vs insertion (node, edge, adj) order
  const int* adj; const float* edge0; const float* nodef;
  if (in_sizes[0] == BN * IN_) {
    nodef = (const float*)d_in[0]; edge0 = (const float*)d_in[1]; adj = (const int*)d_in[2];
  } else {
    adj = (const int*)d_in[0]; edge0 = (const float*)d_in[1]; nodef = (const float*)d_in[2];
  }

  int pi = 3;
  auto nxt = [&]() -> const float* { return (const float*)d_in[pi++]; };

  struct Lin { const float* b; const float* w; };
  struct LNP { const float* b; const float* g; };
  struct Lay {
    const float* attn_matrix;
    Lin edge1, edge2;
    const float* gamma;
    LNP ln1, ln2, ln_eo, ln_out;
    Lin mha_in, mha_out, mlp1, mlp2;
    LNP mlp_ln;
    Lin node_proj;
    LNP norm;
    Lin out_proj, res_proj;
  } Ls[3];
  for (int i = 0; i < NL; i++) {
    Lay& P = Ls[i];
    P.attn_matrix = nxt();
    P.edge1.b = nxt(); P.edge1.w = nxt();
    P.edge2.b = nxt(); P.edge2.w = nxt();
    P.gamma = nxt();
    P.ln1.b = nxt(); P.ln1.g = nxt();
    P.ln2.b = nxt(); P.ln2.g = nxt();
    P.ln_eo.b = nxt(); P.ln_eo.g = nxt();
    P.ln_out.b = nxt(); P.ln_out.g = nxt();
    P.mha_in.b = nxt(); P.mha_in.w = nxt();
    P.mha_out.b = nxt(); P.mha_out.w = nxt();
    P.mlp1.b = nxt(); P.mlp1.w = nxt();
    P.mlp2.b = nxt(); P.mlp2.w = nxt();
    P.mlp_ln.b = nxt(); P.mlp_ln.g = nxt();
    P.node_proj.b = nxt(); P.node_proj.w = nxt();
    P.norm.b = nxt(); P.norm.g = nxt();
    P.out_proj.b = nxt(); P.out_proj.w = nxt();
    if (i == 0) { P.res_proj.b = nxt(); P.res_proj.w = nxt(); }
    else { P.res_proj.b = nullptr; P.res_proj.w = nullptr; }
  }
  Lin pat, pp1, pp2; LNP pln;
  pat.b = nxt(); pat.w = nxt();
  pp1.b = nxt(); pp1.w = nxt();
  pp2.b = nxt(); pp2.w = nxt();
  pln.b = nxt(); pln.g = nxt();

  // workspace arena
  char* wp = (char*)d_ws;
  auto alloc = [&](size_t bytes) -> void* {
    void* p = (void*)wp;
    wp += (bytes + 255) & ~(size_t)255;
    return p;
  };
  float* e_ws      = (float*)alloc(BNN * E_ * 4);
  float* ee        = (float*)alloc(BNN * 4);
  float* xn        = (float*)alloc((size_t)BN * 128 * 4);
  float* node      = (float*)alloc((size_t)BN * 128 * 4);
  _Float16* node_h = (_Float16*)alloc((size_t)BN * 128 * 2);
  float* msg       = (float*)alloc((size_t)BN * 128 * 4);
  float* resb      = (float*)alloc((size_t)BN * 128 * 4);
  float* x2        = (float*)alloc((size_t)BN * 128 * 4);
  float* tmp1      = (float*)alloc((size_t)BN * 128 * 4);
  float* fusedb    = (float*)alloc((size_t)BN * 128 * 4);
  float* hbuf      = (float*)alloc((size_t)BN * 128 * 4);
  float* qkv       = (float*)alloc((size_t)BN * 384 * 4);
  _Float16* qkv_h  = (_Float16*)alloc((size_t)BN * 384 * 2);
  float* obuf      = (float*)alloc((size_t)BN * 128 * 4);
  float* gbuf      = (float*)alloc((size_t)BN * 128 * 4);
  float* xA        = (float*)alloc((size_t)BN * 128 * 4);
  float* xB        = (float*)alloc((size_t)BN * 128 * 4);
  float* rbuf      = (float*)alloc((size_t)BN * 4 * 4);
  float* cbuf      = (float*)alloc((size_t)BN * 4 * 4);
  float* asum      = (float*)alloc(512 * 4);
  float* sbuf      = (float*)alloc((size_t)BN * 4);
  float* pooled    = (float*)alloc(B_ * 128 * 4);
  float* pt1       = (float*)alloc(B_ * 128 * 4);
  float* pt2       = (float*)alloc(B_ * 128 * 4);

  auto gemm = [&](const float* X, Lin lin, const float* res, float* Y,
                  int M, int K, int Nout, int act, int addres) {
    dim3 grid((M + GB_M - 1) / GB_M, (Nout + GB_N - 1) / GB_N);
    gemm_k<<<grid, 128, 0, stream>>>(X, lin.w, lin.b, res, Y, M, K, Nout, act, addres);
  };
  auto lnorm = [&](const float* X, float* Y, LNP p, int rows, int D, int act) {
    ln_rows_k<<<(rows + 3) / 4, 128, 0, stream>>>(X, Y, p.g, p.b, rows, D, act);
  };

  const float* x = nodef;
  for (int i = 0; i < NL; i++) {
    Lay& P = Ls[i];
    int inf = (i == 0) ? IN_ : HID_;
    int outf = OUTF;
    int heads = HEADS_[i];
    int hs = outf / heads;
    int useLeaky = (i != NL - 1);

    lnorm(x, xn, P.ln1, BN, inf, 0);
    gemm(xn, P.node_proj, nullptr, node, BN, inf, outf, 0, 0);
    cvt_h_k<<<512, 256, 0, stream>>>(node, node_h, (size_t)BN * outf);
    asum_k<<<1, 512, 0, stream>>>(P.attn_matrix, asum, heads, 2 * hs + E_);
    const float* e_src = (i == 0) ? edge0 : e_ws;
    edge_fused_k<<<(unsigned)(BNN / 64), 128, 0, stream>>>(
        e_src, e_ws, ee, P.ln2.g, P.ln2.b, P.edge1.w, P.edge1.b,
        P.edge2.w, P.edge2.b, P.ln_eo.g, P.ln_eo.b, asum, 2 * hs);
    rc_k<<<(BN * heads + 127) / 128, 128, 0, stream>>>(node, asum, rbuf, cbuf, BN, heads, hs, outf);
    gat_attn_msg_k<<<B_ * heads * (N_ / 16) / 2, 64, 0, stream>>>(
        node_h, rbuf, cbuf, ee, adj, msg, N_, heads, hs, outf);
    const float* resp;
    if (i == 0) { gemm(x, P.res_proj, nullptr, resb, BN, inf, outf, 0, 0); resp = resb; }
    else resp = x;
    gemm(msg, P.out_proj, resp, tmp1, BN, outf, outf, useLeaky ? 2 : 0, 1);
    lnorm(tmp1, x2, P.ln_out, BN, outf, 0);
    gemm(x2, P.mha_in, nullptr, qkv, BN, outf, 3 * outf, 0, 0);
    cvt_h_k<<<512, 256, 0, stream>>>(qkv, qkv_h, (size_t)BN * 3 * outf);
    mha_attn_k<<<B_ * heads * (N_ / 16) / 2, 64, 0, stream>>>(qkv_h, obuf, N_, heads, hs, outf);
    gemm(obuf, P.mha_out, nullptr, gbuf, BN, outf, outf, 0, 0);
    add3_k<<<512, 256, 0, stream>>>(resp, x2, gbuf, P.gamma, tmp1, (size_t)BN * outf);
    lnorm(tmp1, fusedb, P.norm, BN, outf, 0);
    gemm(fusedb, P.mlp1, nullptr, tmp1, BN, outf, outf, 1, 0);
    lnorm(tmp1, hbuf, P.mlp_ln, BN, outf, 0);
    float* xnext = (i % 2 == 0) ? xA : xB;
    gemm(hbuf, P.mlp2, nullptr, xnext, BN, outf, outf, 0, 0);
    x = xnext;
  }

  // pooling head
  pool_score_k<<<(BN + 127) / 128, 128, 0, stream>>>(x, pat.w, pat.b, sbuf, BN, OUTF);
  pool_softmax_k<<<B_, 32, 0, stream>>>(sbuf, N_);
  pool_weighted_k<<<(B_ * OUTF + 127) / 128, 128, 0, stream>>>(sbuf, x, pooled, B_, N_, OUTF);
  gemm(pooled, pp1, nullptr, pt1, B_, OUTF, 128, 0, 0);
  lnorm(pt1, pt2, pln, B_, 128, 3);               // LN + exact gelu
  gemm(pt2, pp2, nullptr, (float*)d_out, B_, 128, OUTF, 0, 0);
}
